// ScaleDotProductAttention_56229711839584
// MI455X (gfx1250) — matrix operand order repaired and checked by reading, not compiled, not run on hardware
//
#include <hip/hip_runtime.h>

typedef _Float16 f16;
typedef __attribute__((ext_vector_type(16))) _Float16 v16h;
typedef __attribute__((ext_vector_type(8)))  _Float16 v8h;
typedef __attribute__((ext_vector_type(4)))  _Float16 v4h;
typedef __attribute__((ext_vector_type(8)))  float    v8f;
typedef __attribute__((ext_vector_type(4)))  float    v4f;

#define SEQ  4096
#define HD   64
#define NBH  16      // B*H
#define LDSS 72      // padded LDS row stride (f16): 144B, 16B-aligned, conflict-free

static __device__ __forceinline__ v16h cat16(v8h lo, v8h hi) {
    return __builtin_shufflevector(lo, hi, 0,1,2,3,4,5,6,7,8,9,10,11,12,13,14,15);
}

// One 16B global->LDS async copy (GVS mode: SGPR base + VGPR byte offset).
static __device__ __forceinline__ void async_cp16(const f16* lds_dst, unsigned gl_off_bytes,
                                                  const f16* gl_base) {
    unsigned l = (unsigned)(size_t)lds_dst;    // low 32 bits of flat LDS addr = LDS offset
    asm volatile("global_load_async_to_lds_b128 %0, %1, %2"
                 :: "v"(l), "v"(gl_off_bytes), "s"((unsigned long long)(size_t)gl_base)
                 : "memory");
}

// ---------------- prep: K (f32) -> Kh (f16, same layout) ----------------
__global__ __launch_bounds__(256)
void prep_k16(const float* __restrict__ K, f16* __restrict__ Kh) {
    size_t i = ((size_t)blockIdx.x * 256 + threadIdx.x) * 4;
    v4f x = *(const v4f*)(K + i);
    v4h h; h[0] = (f16)x[0]; h[1] = (f16)x[1]; h[2] = (f16)x[2]; h[3] = (f16)x[3];
    *(v4h*)(Kh + i) = h;
}

// ---------------- prep: V (f32 [bh][s][d]) -> Vt (f16 [bh][d][s]) ----------------
__global__ __launch_bounds__(256)
void prep_vt16(const float* __restrict__ V, f16* __restrict__ Vt) {
    __shared__ __align__(16) f16 tb[HD * LDSS];
    const int tid = threadIdx.x;
    const int bh  = blockIdx.y;
    const int s0  = blockIdx.x * 64;
    const float* vsrc = V + ((size_t)bh * SEQ + s0) * HD;
    #pragma unroll
    for (int it = 0; it < 4; ++it) {                // 64x64 tile, coalesced float4 reads
        int idx = (tid + 256 * it) * 4;
        int s = idx >> 6, d = idx & 63;
        v4f x = *(const v4f*)(vsrc + idx);
        tb[(d + 0) * LDSS + s] = (f16)x[0];
        tb[(d + 1) * LDSS + s] = (f16)x[1];
        tb[(d + 2) * LDSS + s] = (f16)x[2];
        tb[(d + 3) * LDSS + s] = (f16)x[3];
    }
    __syncthreads();
    f16* vdst = Vt + (size_t)bh * HD * SEQ + s0;
    #pragma unroll
    for (int it = 0; it < 2; ++it) {                // coalesced 16B writes
        int c = tid + 256 * it;                     // 512 chunks of 8 f16
        int d = c >> 3, j = (c & 7) * 8;
        *(v8h*)(vdst + (size_t)d * SEQ + j) = *(const v8h*)(&tb[d * LDSS + j]);
    }
}

// ---------------- fused causal flash attention ----------------
// Block = 8 waves = 256 threads; wave w owns 16 query rows; block owns 128.
// Orientation: S^T = K_tile x Q^T, O^T = V^T x P^T -> per-lane scalar softmax state.
__global__ __launch_bounds__(256, 1)
void fa_fwd_causal(const float* __restrict__ Q, const f16* __restrict__ Kh,
                   const f16* __restrict__ Vt, float* __restrict__ O)
{
    __shared__ __align__(16) f16 sK [2][64 * LDSS];   // K tile  [key][d], double-buffered
    __shared__ __align__(16) f16 sVT[2][HD * LDSS];   // V^T tile [d][key], double-buffered
    __shared__ __align__(16) f16 sP [8][16 * LDSS];   // per-wave P^T staging [q][key]

    const int tid  = threadIdx.x;
    const int wave = tid >> 5;
    const int lane = tid & 31;
    const int n    = lane & 15;
    const int half = lane >> 4;

    const int qtile = (int)(gridDim.x - 1) - (int)blockIdx.x;   // heavy blocks first
    const int bh    = (int)blockIdx.z * 8 + (int)blockIdx.y;
    const size_t base = (size_t)bh * SEQ * HD;

    const int qb = qtile * 128 + wave * 16;     // wave's first query row
    const int q  = qb + n;                      // this lane's query row

    // ---- Q row into B-layout regs, pre-scaled to log2 domain ----
    // B 32x16 layout: lane col n = lane%16; element i -> k(d) = 32c + 16*half + i
    const float sc = 0.125f * 1.44269504088896340736f;   // (1/sqrt(64))*log2(e)
    v16h bq[2];
    {
        const float* qrow = Q + base + (size_t)q * HD;
        #pragma unroll
        for (int c = 0; c < 2; ++c)
            #pragma unroll
            for (int i = 0; i < 16; ++i)
                bq[c][i] = (f16)(qrow[c * 32 + 16 * half + i] * sc);
    }

    v8f acc[4];                                  // O^T: acc[t][r] = O[q][t*16 + 8*half + r]
    #pragma unroll
    for (int t = 0; t < 4; ++t)
        #pragma unroll
        for (int r = 0; r < 8; ++r) acc[t][r] = 0.0f;
    float rmax = -1e30f, rsum = 0.0f;            // per-lane scalars (one query each)

    const f16* kgb = Kh + base;                          // tile kt at + kt*64*HD
    const f16* vgb = Vt + (size_t)bh * HD * SEQ;         // row d stride SEQ, tile col kt*64

    const int nkt = 2 * qtile + 2;

    // ---- async prefill of tile 0 (buffer 0): 4x16B per thread ----
    #pragma unroll
    for (int s2 = 0; s2 < 2; ++s2) {
        int e = tid * 8 + 2048 * s2;
        int r0 = e >> 6, c0 = e & 63;
        async_cp16(&sK [0][r0 * LDSS + c0], (unsigned)(e * 2), kgb);
        async_cp16(&sVT[0][r0 * LDSS + c0], (unsigned)((r0 * SEQ + c0) * 2), vgb);
    }

    for (int kt = 0; kt < nkt; ++kt) {
        const int buf = kt & 1;
        asm volatile("s_wait_asynccnt 0" ::: "memory");   // my fills for tile kt done
        __syncthreads();                                   // everyone's done + old buf free
        if (kt + 1 < nkt) {                                // async fill of next tile
            const f16* kg = kgb + (size_t)(kt + 1) * 64 * HD;
            const f16* vg = vgb + (size_t)(kt + 1) * 64;
            #pragma unroll
            for (int s2 = 0; s2 < 2; ++s2) {
                int e = tid * 8 + 2048 * s2;
                int r0 = e >> 6, c0 = e & 63;
                async_cp16(&sK [buf ^ 1][r0 * LDSS + c0], (unsigned)(e * 2), kg);
                async_cp16(&sVT[buf ^ 1][r0 * LDSS + c0], (unsigned)((r0 * SEQ + c0) * 2), vg);
            }
        }
        if (kt * 64 > qb + 15) continue;           // no valid keys for this wave's rows

        const f16* sKb = sK[buf];
        const f16* sVb = sVT[buf];
        f16* pw = sP[wave];
        const bool diag = (kt * 64 + 63 > qb);     // only diagonal tiles need masking

        // ---- S^T: 4 M-tiles (16 keys each) x 2 d-chunks ----
        // A 16x32 layout: lane row = mt*16 + n; element i -> d = 32c + 16*half + i
        v8f s[4];
        #pragma unroll
        for (int mt = 0; mt < 4; ++mt) {
            v8f z;
            #pragma unroll
            for (int r = 0; r < 8; ++r) z[r] = 0.0f;
            #pragma unroll
            for (int c = 0; c < 2; ++c) {
                const f16* src = &sKb[(mt * 16 + n) * LDSS + c * 32 + 16 * half];
                v16h a = cat16(*(const v8h*)src, *(const v8h*)(src + 8));
                z = __builtin_amdgcn_wmma_f32_16x16x32_f16(false, a, false, bq[c],
                                                           (short)0, z, false, false);
            }
            s[mt] = z;
        }
        if (diag) {                                 // C-layout: s[mt][r] is key kt*64+mt*16+8*half+r
            #pragma unroll
            for (int mt = 0; mt < 4; ++mt) {
                int kb = kt * 64 + mt * 16 + 8 * half;
                #pragma unroll
                for (int r = 0; r < 8; ++r)
                    s[mt][r] = (kb + r <= q) ? s[mt][r] : -1e30f;
            }
        }

        // ---- online softmax: local 32-value tree + one cross-half shuffle ----
        float tmax = rmax;
        #pragma unroll
        for (int mt = 0; mt < 4; ++mt)
            #pragma unroll
            for (int r = 0; r < 8; ++r) tmax = fmaxf(tmax, s[mt][r]);
        tmax = fmaxf(tmax, __shfl_xor(tmax, 16, 32));
        const float alpha = exp2f(rmax - tmax);
        rmax = tmax;

        float psum = 0.0f;
        #pragma unroll
        for (int mt = 0; mt < 4; ++mt) {            // exp, pack 8 consecutive keys, 16B store
            v8h ph;
            #pragma unroll
            for (int r = 0; r < 8; ++r) {
                float p = exp2f(s[mt][r] - tmax);
                psum += p;
                ph[r] = (f16)p;
            }
            *(v8h*)(&pw[n * LDSS + mt * 16 + 8 * half]) = ph;
        }
        psum += __shfl_xor(psum, 16, 32);
        rsum = rsum * alpha + psum;
        #pragma unroll
        for (int t = 0; t < 4; ++t)
            #pragma unroll
            for (int r = 0; r < 8; ++r) acc[t][r] *= alpha;

        // ---- P^T in B-layout (wave-internal DS ordering, no barrier needed) ----
        v16h bp[2];
        #pragma unroll
        for (int c = 0; c < 2; ++c) {
            const f16* ps = &pw[n * LDSS + c * 32 + 16 * half];
            bp[c] = cat16(*(const v8h*)ps, *(const v8h*)(ps + 8));
        }
        // ---- O^T += V^T x P^T ----
        #pragma unroll
        for (int t = 0; t < 4; ++t)
            #pragma unroll
            for (int c = 0; c < 2; ++c) {
                const f16* vs = &sVb[(t * 16 + n) * LDSS + c * 32 + 16 * half];
                v16h a = cat16(*(const v8h*)vs, *(const v8h*)(vs + 8));
                acc[t] = __builtin_amdgcn_wmma_f32_16x16x32_f16(false, a, false, bp[c],
                                                                (short)0, acc[t], false, false);
            }
    }

    // ---- epilogue: contiguous 16B stores per t-tile ----
    const float inv = 1.0f / rsum;
    float* orow = O + base + (size_t)q * HD;
    #pragma unroll
    for (int t = 0; t < 4; ++t) {
        v4f o0, o1;
        #pragma unroll
        for (int r = 0; r < 4; ++r) { o0[r] = acc[t][r] * inv; o1[r] = acc[t][r + 4] * inv; }
        *(v4f*)(orow + t * 16 + 8 * half)     = o0;
        *(v4f*)(orow + t * 16 + 8 * half + 4) = o1;
    }
}

extern "C" void kernel_launch(void* const* d_in, const int* in_sizes, int n_in,
                              void* d_out, int out_size, void* d_ws, size_t ws_size,
                              hipStream_t stream)
{
    const float* q = (const float*)d_in[0];
    const float* k = (const float*)d_in[1];
    const float* v = (const float*)d_in[2];
    float* out = (float*)d_out;

    const size_t nel = (size_t)NBH * SEQ * HD;   // 4,194,304 elements per tensor
    f16* Kh = (f16*)d_ws;                        // 8 MB
    f16* Vt = Kh + nel;                          // 8 MB  (needs ws_size >= 16 MB)

    prep_k16 <<<(unsigned)(nel / 1024), 256, 0, stream>>>(k, Kh);
    prep_vt16<<<dim3(SEQ / 64, NBH), 256, 0, stream>>>(v, Vt);
    fa_fwd_causal<<<dim3(SEQ / 128, 8, 2), 256, 0, stream>>>(q, Kh, Vt, out);
}
